// SplatsVAE_76098230550652
// MI455X (gfx1250) — compile-verified, matrix operand order checked by reading, not run on hardware
//
#include <hip/hip_runtime.h>

typedef __attribute__((ext_vector_type(8)))  _Float16 v8h;
typedef __attribute__((ext_vector_type(16))) _Float16 v16h;
typedef __attribute__((ext_vector_type(8)))  float    v8f;

#define BSZ   8
#define DDIM  64
#define HIDN  256
#define KSPL  2048
#define IMGS  256

// ---------------------------------------------------------------------------
// Generic dense layer: Y[M,N] = act(X[M,Kd] @ W[Kd,N] + bias[N])
// Tiny (<= 17 M MAC total) -> plain VALU, coalesced on W columns.
// ---------------------------------------------------------------------------
__global__ void splats_mlp_kernel(const float* __restrict__ X,
                                  const float* __restrict__ W,
                                  const float* __restrict__ bias,
                                  float* __restrict__ Y,
                                  int M, int Kd, int N, int do_relu) {
    int idx = blockIdx.x * blockDim.x + threadIdx.x;
    if (idx >= M * N) return;
    int m = idx / N, n = idx - m * N;
    const float* x = X + (size_t)m * Kd;
    const float* w = W + n;
    float acc = bias[n];
    #pragma unroll 4
    for (int k = 0; k < Kd; ++k) acc = fmaf(x[k], w[(size_t)k * N], acc);
    if (do_relu) acc = fmaxf(acc, 0.f);
    Y[idx] = acc;
}

// ---------------------------------------------------------------------------
// Separable Gaussian profiles.
//   ExT[b][p][k] = exp(-(grid[p]-xc)^2 / (2 r^2))                (f16)
//   EyT[b][p][k] = amp * exp(-(grid[p]-yc)^2 / (2 r^2))          (f16)
// stored (pixel, K) row-major so both WMMA fragments load identically.
// Stores are coalesced across k (thread index).
// ---------------------------------------------------------------------------
__global__ void splats_profile_kernel(const float* __restrict__ geom,
                                      const float* __restrict__ app,
                                      _Float16* __restrict__ ExT,
                                      _Float16* __restrict__ EyT) {
    int idx = blockIdx.x * blockDim.x + threadIdx.x;     // b*KSPL + k
    if (idx >= BSZ * KSPL) return;
    int b = idx / KSPL, k = idx - b * KSPL;

    const float* gm = geom + (size_t)idx * 4;
    const float* ap = app  + (size_t)idx * 2;
    float xc = gm[0], yc = gm[1], rraw = gm[3];
    // softplus then clip(min=0.01)
    float sp  = (rraw > 20.f) ? rraw : log1pf(__expf(rraw));
    float r   = fmaxf(sp, 0.01f);
    float inv = 1.f / (2.f * r * r);
    float amp = (1.f / (1.f + __expf(-ap[0]))) *
                (1.f / (1.f + __expf(-ap[1])));          // intensity * opacity

    _Float16* ex = ExT + (size_t)b * IMGS * KSPL + k;
    _Float16* ey = EyT + (size_t)b * IMGS * KSPL + k;
    #pragma unroll 4
    for (int p = 0; p < IMGS; ++p) {
        float gp = -1.f + (2.f / 255.f) * (float)p;
        float dx = gp - xc;
        float dy = gp - yc;
        ex[(size_t)p * KSPL] = (_Float16)__expf(-dx * dx * inv);
        ey[(size_t)p * KSPL] = (_Float16)(amp * __expf(-dy * dy * inv));
    }
}

// ---------------------------------------------------------------------------
// Rendering GEMM: img[b] = EyT(256 x 2048) @ ExT(256 x 2048)^T, clipped [0,1].
// One wave computes a 64x64 tile via 4x4 v_wmma_f32_16x16x32_f16 tiles.
//
// 16-bit A fragment layout (ISA 7.12.2): lane L in [0,16): row M = m0+L,
// VGPRs 0-3 hold K = k0+0..7, VGPRs 4-7 hold K = k0+16..23; lanes 16-31 hold
// the same rows at K offsets +8 / +24.  B (consumed as N x K here) mirrors it.
// ---------------------------------------------------------------------------
__device__ __forceinline__ v16h splats_load_frag(const _Float16* base) {
    v8h lo = *(const v8h*)(base);        // K = +0 .. +7   -> VGPRs 0-3
    v8h hi = *(const v8h*)(base + 16);   // K = +16 .. +23 -> VGPRs 4-7
    return __builtin_shufflevector(lo, hi, 0, 1, 2, 3, 4, 5, 6, 7,
                                           8, 9, 10, 11, 12, 13, 14, 15);
}

__global__ void __launch_bounds__(256)
splats_render_wmma_kernel(const _Float16* __restrict__ EyT,
                          const _Float16* __restrict__ ExT,
                          float* __restrict__ out) {
    const int lane = threadIdx.x & 31;
    const int wave = threadIdx.x >> 5;
    const int half = lane >> 4;          // 0: K-lo half, 1: K-hi half
    const int row  = lane & 15;

    // 128 64x64 tiles total: tile = (b, m-tile, n-tile)
    int tile = blockIdx.x * 8 + wave;    // gridDim.x = 16
    int b    = tile >> 4;
    int rem  = tile & 15;
    int m0   = (rem >> 2) * 64;          // image row (y) tile
    int n0   = (rem & 3) * 64;           // image col (x) tile

    const _Float16* A  = EyT + (size_t)b * IMGS * KSPL;   // M x K (rows = h)
    const _Float16* Bm = ExT + (size_t)b * IMGS * KSPL;   // N x K (rows = w)

    v8f zero = {};
    v8f c[4][4];
    #pragma unroll
    for (int i = 0; i < 4; ++i)
        #pragma unroll
        for (int j = 0; j < 4; ++j) c[i][j] = zero;

    const int koff = half * 8;
    for (int k0 = 0; k0 < KSPL; k0 += 32) {
        v16h af[4], bf[4];
        #pragma unroll
        for (int i = 0; i < 4; ++i)
            af[i] = splats_load_frag(A + (size_t)(m0 + 16 * i + row) * KSPL + k0 + koff);
        #pragma unroll
        for (int j = 0; j < 4; ++j)
            bf[j] = splats_load_frag(Bm + (size_t)(n0 + 16 * j + row) * KSPL + k0 + koff);
        #pragma unroll
        for (int i = 0; i < 4; ++i)
            #pragma unroll
            for (int j = 0; j < 4; ++j)
                c[i][j] = __builtin_amdgcn_wmma_f32_16x16x32_f16(
                    /*neg_a=*/false, af[i], /*neg_b=*/false, bf[j],
                    /*c_mod=*/(short)0, c[i][j],
                    /*reuse_a=*/false, /*reuse_b=*/false);
    }

    // C/D layout: VGPR v, lanes 0-15 -> (M = v,   N = lane);
    //             lanes 16-31        -> (M = v+8, N = lane-16)
    float* o = out + (size_t)b * IMGS * IMGS;
    #pragma unroll
    for (int i = 0; i < 4; ++i)
        #pragma unroll
        for (int j = 0; j < 4; ++j)
            #pragma unroll
            for (int v = 0; v < 8; ++v) {
                int h = m0 + 16 * i + half * 8 + v;
                int w = n0 + 16 * j + row;
                float val = c[i][j][v];
                o[(size_t)h * IMGS + w] = fminf(fmaxf(val, 0.f), 1.f);
            }
}

// ---------------------------------------------------------------------------
extern "C" void kernel_launch(void* const* d_in, const int* in_sizes, int n_in,
                              void* d_out, int out_size, void* d_ws, size_t ws_size,
                              hipStream_t stream) {
    const float* z   = (const float*)d_in[0];
    const float* Wg1 = (const float*)d_in[1];
    const float* bg1 = (const float*)d_in[2];
    const float* Wg2 = (const float*)d_in[3];
    const float* bg2 = (const float*)d_in[4];
    const float* Wg3 = (const float*)d_in[5];
    const float* bg3 = (const float*)d_in[6];
    const float* Wa1 = (const float*)d_in[7];
    const float* ba1 = (const float*)d_in[8];
    const float* Wa2 = (const float*)d_in[9];
    const float* ba2 = (const float*)d_in[10];
    const float* Wa3 = (const float*)d_in[11];
    const float* ba3 = (const float*)d_in[12];
    float* out = (float*)d_out;

    // Workspace carve-up (bytes)
    char* ws = (char*)d_ws;
    size_t off = 0;
    float* g1   = (float*)(ws + off); off += (size_t)BSZ * HIDN * 4;        // 8K
    float* g2   = (float*)(ws + off); off += (size_t)BSZ * HIDN * 4;
    float* a1   = (float*)(ws + off); off += (size_t)BSZ * HIDN * 4;
    float* a2   = (float*)(ws + off); off += (size_t)BSZ * HIDN * 4;
    float* geom = (float*)(ws + off); off += (size_t)BSZ * KSPL * 4 * 4;    // 256K
    float* app  = (float*)(ws + off); off += (size_t)BSZ * KSPL * 2 * 4;    // 128K
    _Float16* ExT = (_Float16*)(ws + off); off += (size_t)BSZ * IMGS * KSPL * 2; // 8M
    _Float16* EyT = (_Float16*)(ws + off); off += (size_t)BSZ * IMGS * KSPL * 2; // 8M
    (void)ws_size; (void)in_sizes; (void)n_in; (void)out_size;

    const int T = 256;
    // Geometry path MLP
    splats_mlp_kernel<<<(BSZ * HIDN + T - 1) / T, T, 0, stream>>>(z,  Wg1, bg1, g1,   BSZ, DDIM, HIDN,     1);
    splats_mlp_kernel<<<(BSZ * HIDN + T - 1) / T, T, 0, stream>>>(g1, Wg2, bg2, g2,   BSZ, HIDN, HIDN,     1);
    splats_mlp_kernel<<<(BSZ * KSPL * 4 + T - 1) / T, T, 0, stream>>>(g2, Wg3, bg3, geom, BSZ, HIDN, KSPL * 4, 0);
    // Appearance path MLP
    splats_mlp_kernel<<<(BSZ * HIDN + T - 1) / T, T, 0, stream>>>(z,  Wa1, ba1, a1,   BSZ, DDIM, HIDN,     1);
    splats_mlp_kernel<<<(BSZ * HIDN + T - 1) / T, T, 0, stream>>>(a1, Wa2, ba2, a2,   BSZ, HIDN, HIDN,     1);
    splats_mlp_kernel<<<(BSZ * KSPL * 2 + T - 1) / T, T, 0, stream>>>(a2, Wa3, ba3, app,  BSZ, HIDN, KSPL * 2, 0);

    // Separable Gaussian profile generation (8.4M exps instead of 1.07G)
    splats_profile_kernel<<<(BSZ * KSPL + T - 1) / T, T, 0, stream>>>(geom, app, ExT, EyT);

    // Rank-2048 outer-product accumulation as f16 WMMA GEMM + clip
    splats_render_wmma_kernel<<<16, 256, 0, stream>>>(EyT, ExT, out);
}
